// ConditionalPointerHead_42082089566679
// MI455X (gfx1250) — matrix operand order repaired and checked by reading, not compiled, not run on hardware
//
#include <hip/hip_runtime.h>
#include <math.h>

// Problem constants (from reference): B=8, L=4096, d=768, e=1536, TOPK=10
#define BB 8
#define LL 4096
#define DD 768
#define EE 1536
#define NTOPK 10
#define MASK_VAL -100.0f
#define SCALE_E 0.025515518153991442f  // 1/sqrt(1536)

typedef __attribute__((ext_vector_type(16))) __bf16 v16bf;
typedef __attribute__((ext_vector_type(8)))  float  v8f;

union BF16x16 { v16bf v; unsigned short s[16]; };

__device__ __forceinline__ unsigned short f2bf(float x) {
  unsigned int u = __float_as_uint(x);
  unsigned int r = u + 0x7FFFu + ((u >> 16) & 1u);  // round-to-nearest-even
  return (unsigned short)(r >> 16);
}
__device__ __forceinline__ float bf2f(unsigned short h) {
  return __uint_as_float(((unsigned int)h) << 16);
}

// Split 16 fp32 into hi/lo bf16 fragments (x ~= hi + lo).
__device__ __forceinline__ void split16(const float* f, v16bf& fh, v16bf& fl) {
  BF16x16 Hh, Ll;
#pragma unroll
  for (int j = 0; j < 16; j++) {
    unsigned short h = f2bf(f[j]);
    Hh.s[j] = h;
    Ll.s[j] = f2bf(f[j] - bf2f(h));
  }
  fh = Hh.v; fl = Ll.v;
}

// A-fragment (16x32, MxK) from a row-major fp32 row.
// ISA 16-bit A layout: lanes 0-15 (hi=0): K = kb+{0..7, 16..23}; lanes 16-31: +8.
__device__ __forceinline__ void load_frag_a(const float* __restrict__ row, int kb, int hi,
                                            v16bf& fh, v16bf& fl) {
  const float4* p0 = reinterpret_cast<const float4*>(row + kb + hi * 8);
  const float4* p1 = reinterpret_cast<const float4*>(row + kb + 16 + hi * 8);
  float4 a0 = p0[0], a1 = p0[1], b0 = p1[0], b1 = p1[1];
  float f[16] = {a0.x,a0.y,a0.z,a0.w, a1.x,a1.y,a1.z,a1.w,
                 b0.x,b0.y,b0.z,b0.w, b1.x,b1.y,b1.z,b1.w};
  split16(f, fh, fl);
}

// B-fragment (32x16, KxN) where B[k][n] = src_row_n[k] (src row-major over k).
// B layout (matches A): lanes 0-15 K=kb+{0..7, 16..23}; lanes 16-31: +8.
__device__ __forceinline__ void load_frag_b(const float* __restrict__ row, int kb, int hi,
                                            v16bf& fh, v16bf& fl) {
  const float4* p0 = reinterpret_cast<const float4*>(row + kb + hi * 8);
  const float4* p1 = reinterpret_cast<const float4*>(row + kb + 16 + hi * 8);
  float4 a = p0[0], b = p0[1], c = p1[0], d = p1[1];
  float f[16] = {a.x,a.y,a.z,a.w, b.x,b.y,b.z,b.w,
                 c.x,c.y,c.z,c.w, d.x,d.y,d.z,d.w};
  split16(f, fh, fl);
}

__device__ __forceinline__ v8f wmma_bf16(v16bf a, v16bf b, v8f c) {
  return __builtin_amdgcn_wmma_f32_16x16x32_bf16(false, a, false, b, (short)0, c, false, false);
}
// fp32-ish product via 3 bf16 WMMAs: ah*bh + ah*bl + al*bh
__device__ __forceinline__ v8f mma3(v16bf ah, v16bf al, v16bf bh, v16bf bl, v8f c) {
  c = wmma_bf16(ah, bh, c);
  c = wmma_bf16(ah, bl, c);
  c = wmma_bf16(al, bh, c);
  return c;
}

// ---------------------------------------------------------------------------
// Kernel: repack H [32768][768] fp32 -> A-fragment tiled hi/lo bf16.
// Tile index = mt*24 + kt (mt: 16-row tile, kt: 32-k tile); 512 bf16 per tile.
__global__ void repack_H_kernel(const float* __restrict__ H,
                                unsigned short* __restrict__ Ahi,
                                unsigned short* __restrict__ Alo) {
  size_t t = (size_t)blockIdx.x * 256 + threadIdx.x;  // 2048*24*32 threads
  int lane = (int)(t & 31);
  size_t tile = t >> 5;
  int kt = (int)(tile % 24);
  int mt = (int)(tile / 24);
  int m = lane & 15, hi = lane >> 4;
  const float* row = H + ((size_t)(mt * 16 + m)) * DD + kt * 32;
  const float4* p0 = reinterpret_cast<const float4*>(row + hi * 8);
  const float4* p1 = reinterpret_cast<const float4*>(row + 16 + hi * 8);
  float4 a0 = p0[0], a1 = p0[1], b0 = p1[0], b1 = p1[1];
  float f[16] = {a0.x,a0.y,a0.z,a0.w, a1.x,a1.y,a1.z,a1.w,
                 b0.x,b0.y,b0.z,b0.w, b1.x,b1.y,b1.z,b1.w};
  size_t dst = tile * 512 + (size_t)lane * 16;
#pragma unroll
  for (int j = 0; j < 16; j++) {
    unsigned short h = f2bf(f[j]);
    Ahi[dst + j] = h;
    Alo[dst + j] = f2bf(f[j] - bf2f(h));
  }
}

// Kernel: repack W_q/W_k/W_v/W_c [768][1536] fp32 -> B-fragment tiled hi/lo bf16.
// Per matrix: tiles (nt*24 + kt), nt in [0,96), kt in [0,24); 512 bf16 per tile.
__global__ void repack_W_kernel(const float* __restrict__ W0, const float* __restrict__ W1,
                                const float* __restrict__ W2, const float* __restrict__ W3,
                                unsigned short* __restrict__ Whi,
                                unsigned short* __restrict__ Wlo) {
  size_t t = (size_t)blockIdx.x * 256 + threadIdx.x;  // 4*2304*32 threads
  int lane = (int)(t & 31);
  size_t tile = t >> 5;                 // [0, 4*2304)
  int mat = (int)(tile / 2304);
  int r = (int)(tile % 2304);
  int nt = r / 24, kt = r % 24;
  const float* W = (mat == 0) ? W0 : (mat == 1) ? W1 : (mat == 2) ? W2 : W3;
  int nlo = lane & 15, hi = lane >> 4;
  int col = nt * 16 + nlo;
  size_t dst = tile * 512 + (size_t)lane * 16;
#pragma unroll
  for (int j = 0; j < 16; j++) {
    int k = kt * 32 + (j < 8 ? hi * 8 + j : 16 + hi * 8 + (j - 8));      // B layout: table order, matches A
    float v = W[(size_t)k * EE + col];
    unsigned short h = f2bf(v);
    Whi[dst + j] = h;
    Wlo[dst + j] = f2bf(v - bf2f(h));
  }
}

// ---------------------------------------------------------------------------
// Kernel: start_logits[b,l] = H[b,l,:].W_start + b_start, masked. One wave/token.
__global__ void start_kernel(const float* __restrict__ H, const int* __restrict__ amask,
                             const float* __restrict__ Ws, const float* __restrict__ bs,
                             float* __restrict__ out) {
  int lane = threadIdx.x & 31, wave = threadIdx.x >> 5;
  size_t row = (size_t)blockIdx.x * 8 + wave;  // 32768 rows
  const float* h = H + row * DD;
  float acc = 0.f;
  for (int i = lane; i < DD; i += 32) acc += h[i] * Ws[i];
  for (int off = 16; off > 0; off >>= 1) acc += __shfl_xor(acc, off, 32);
  if (lane == 0) {
    float v = acc + bs[0];
    out[row] = (amask[row] != 0) ? v : MASK_VAL;
  }
}

// Kernel: per-batch softmax stats + iterative top-10 + renormalized weights.
__global__ void topk_kernel(const float* __restrict__ logits,
                            int* __restrict__ tidx, float* __restrict__ twt) {
  int b = blockIdx.x, tid = threadIdx.x;
  const float* row = logits + (size_t)b * LL;
  __shared__ float sval[256];
  __shared__ int   sidx[256];
  __shared__ int   chosen[NTOPK];
  __shared__ float chval[NTOPK];

  float m = -1e30f;
  for (int i = tid; i < LL; i += 256) m = fmaxf(m, row[i]);
  sval[tid] = m; __syncthreads();
  for (int s = 128; s > 0; s >>= 1) { if (tid < s) sval[tid] = fmaxf(sval[tid], sval[tid + s]); __syncthreads(); }
  float rmax = sval[0]; __syncthreads();

  float z = 0.f;
  for (int i = tid; i < LL; i += 256) z += expf(row[i] - rmax);
  sval[tid] = z; __syncthreads();
  for (int s = 128; s > 0; s >>= 1) { if (tid < s) sval[tid] += sval[tid + s]; __syncthreads(); }
  float Z = sval[0]; __syncthreads();

  for (int t = 0; t < NTOPK; t++) {
    float bv = -1e30f; int bi = 0x7fffffff;
    for (int i = tid; i < LL; i += 256) {
      bool skip = false;
      for (int c = 0; c < t; c++) skip = skip || (chosen[c] == i);
      float v = skip ? -1e30f : row[i];
      if (v > bv || (v == bv && i < bi)) { bv = v; bi = i; }
    }
    sval[tid] = bv; sidx[tid] = bi; __syncthreads();
    for (int s = 128; s > 0; s >>= 1) {
      if (tid < s) {
        if (sval[tid + s] > sval[tid] ||
            (sval[tid + s] == sval[tid] && sidx[tid + s] < sidx[tid])) {
          sval[tid] = sval[tid + s]; sidx[tid] = sidx[tid + s];
        }
      }
      __syncthreads();
    }
    if (tid == 0) { chosen[t] = sidx[0]; chval[t] = sval[0]; }
    __syncthreads();
  }
  if (tid == 0) {
    float p[NTOPK]; float wsum = 0.f;
    for (int t = 0; t < NTOPK; t++) { p[t] = expf(chval[t] - rmax) / Z; wsum += p[t]; }
    for (int t = 0; t < NTOPK; t++) {
      twt[b * NTOPK + t] = p[t] / (wsum + 1e-9f);
      tidx[b * NTOPK + t] = chosen[t];
    }
  }
}

// ---------------------------------------------------------------------------
// Kernel: Q[80][1536] = gather(H, topk_idx) @ W_q + b_q   (WMMA, hi/lo split)
__global__ void gemmQ_kernel(const float* __restrict__ H, const int* __restrict__ tidx,
                             const unsigned short* __restrict__ Bhi,
                             const unsigned short* __restrict__ Blo,
                             const float* __restrict__ bias, float* __restrict__ Q) {
  int lane = threadIdx.x & 31, wave = threadIdx.x >> 5;
  int mt = blockIdx.x;                       // 0..4 (80 rows)
  int nt0 = blockIdx.y * 16 + wave * 2;      // 96 n-tiles total
  int m = lane & 15, hi = lane >> 4;
  int r = mt * 16 + m;                       // < 80 always
  int b = r / NTOPK, kk = r % NTOPK;
  const float* row = H + ((size_t)b * LL + tidx[b * NTOPK + kk]) * DD;
  v8f acc[2] = {};
  for (int kt = 0; kt < 24; kt++) {
    v16bf ah, al; load_frag_a(row, kt * 32, hi, ah, al);
#pragma unroll
    for (int j = 0; j < 2; j++) {
      size_t tb = ((size_t)(nt0 + j) * 24 + kt) * 512 + (size_t)lane * 16;
      v16bf bh = *reinterpret_cast<const v16bf*>(Bhi + tb);
      v16bf bl = *reinterpret_cast<const v16bf*>(Blo + tb);
      acc[j] = mma3(ah, al, bh, bl, acc[j]);
    }
  }
  int nlo = lane & 15;
#pragma unroll
  for (int j = 0; j < 2; j++) {
    int n = (nt0 + j) * 16 + nlo;
    float bn = bias[n];
#pragma unroll
    for (int p = 0; p < 8; p++) {
      int mo = mt * 16 + hi * 8 + p;        // < 80
      Q[(size_t)mo * EE + n] = acc[j][p] + bn;
    }
  }
}

// Kernel: big projection GEMM  out[32768][1536] = H @ W + bias  (WMMA, hi/lo)
// Block: 8 waves as 2(M) x 4(N); wave tile 32x32; block tile 64x128.
__global__ void gemm_big_kernel(const unsigned short* __restrict__ Ahi,
                                const unsigned short* __restrict__ Alo,
                                const unsigned short* __restrict__ Bhi,
                                const unsigned short* __restrict__ Blo,
                                const float* __restrict__ bias, float* __restrict__ out) {
  int lane = threadIdx.x & 31, wave = threadIdx.x >> 5;
  int wm = wave & 1, wn = wave >> 1;
  int mt0 = blockIdx.x * 4 + wm * 2;         // 2048 m-tiles
  int nt0 = blockIdx.y * 8 + wn * 2;         // 96 n-tiles
  v8f acc[2][2] = {};
  for (int kt = 0; kt < 24; kt++) {
    v16bf ah[2], al[2], bh[2], bl[2];
#pragma unroll
    for (int i = 0; i < 2; i++) {
      size_t ta = ((size_t)(mt0 + i) * 24 + kt) * 512 + (size_t)lane * 16;
      ah[i] = *reinterpret_cast<const v16bf*>(Ahi + ta);
      al[i] = *reinterpret_cast<const v16bf*>(Alo + ta);
      if (kt + 1 < 24) __builtin_prefetch(Ahi + ta + 512, 0, 1);
    }
#pragma unroll
    for (int j = 0; j < 2; j++) {
      size_t tb = ((size_t)(nt0 + j) * 24 + kt) * 512 + (size_t)lane * 16;
      bh[j] = *reinterpret_cast<const v16bf*>(Bhi + tb);
      bl[j] = *reinterpret_cast<const v16bf*>(Blo + tb);
    }
#pragma unroll
    for (int i = 0; i < 2; i++)
#pragma unroll
      for (int j = 0; j < 2; j++)
        acc[i][j] = mma3(ah[i], al[i], bh[j], bl[j], acc[i][j]);
  }
  int nlo = lane & 15, hi = lane >> 4;
#pragma unroll
  for (int i = 0; i < 2; i++)
#pragma unroll
    for (int j = 0; j < 2; j++) {
      int n = (nt0 + j) * 16 + nlo;
      float bn = bias[n];
#pragma unroll
      for (int p = 0; p < 8; p++) {
        int mrow = (mt0 + i) * 16 + hi * 8 + p;
        out[(size_t)mrow * EE + n] = acc[i][j][p] + bn;
      }
    }
}

// ---------------------------------------------------------------------------
// Kernel: attn scores = Q @ K_^T * scale, masked.  Per batch; wave does 16x32.
__global__ void scores_kernel(const float* __restrict__ Q, const float* __restrict__ Km,
                              const int* __restrict__ amask, float* __restrict__ attn) {
  int b = blockIdx.x;
  int lane = threadIdx.x & 31, wave = threadIdx.x >> 5;
  int nt0 = blockIdx.y * 16 + wave * 2;      // 256 L-tiles
  int mrow = lane & 15, hi = lane >> 4;
  const float* qrow = Q + (size_t)(b * NTOPK + (mrow < NTOPK ? mrow : NTOPK - 1)) * EE;
  v8f acc[2] = {};
  for (int kt = 0; kt < 48; kt++) {
    v16bf ah, al; load_frag_a(qrow, kt * 32, hi, ah, al);
#pragma unroll
    for (int j = 0; j < 2; j++) {
      int tok = (nt0 + j) * 16 + mrow;       // lane&15 indexes B column
      const float* krow = Km + ((size_t)b * LL + tok) * EE;
      v16bf bh, bl; load_frag_b(krow, kt * 32, hi, bh, bl);
      acc[j] = mma3(ah, al, bh, bl, acc[j]);
    }
  }
#pragma unroll
  for (int j = 0; j < 2; j++) {
    int tok = (nt0 + j) * 16 + (lane & 15);
    bool valid = amask[(size_t)b * LL + tok] != 0;
#pragma unroll
    for (int p = 0; p < 8; p++) {
      int m = hi * 8 + p;
      if (m < NTOPK) {
        float v = acc[j][p] * SCALE_E;
        if (!valid) v = MASK_VAL;
        attn[((size_t)(b * NTOPK + m)) * LL + tok] = v;
      }
    }
  }
}

// Kernel: in-place softmax over each of the 80 attention rows (length 4096).
__global__ void attn_softmax_kernel(float* __restrict__ attn) {
  int r = blockIdx.x, tid = threadIdx.x;
  float* row = attn + (size_t)r * LL;
  __shared__ float s[256];
  float m = -1e30f;
  for (int i = tid; i < LL; i += 256) m = fmaxf(m, row[i]);
  s[tid] = m; __syncthreads();
  for (int k = 128; k > 0; k >>= 1) { if (tid < k) s[tid] = fmaxf(s[tid], s[tid + k]); __syncthreads(); }
  float rmax = s[0]; __syncthreads();
  float z = 0.f;
  for (int i = tid; i < LL; i += 256) z += expf(row[i] - rmax);
  s[tid] = z; __syncthreads();
  for (int k = 128; k > 0; k >>= 1) { if (tid < k) s[tid] += s[tid + k]; __syncthreads(); }
  float Z = s[0]; __syncthreads();
  float inv = 1.f / Z;
  for (int i = tid; i < LL; i += 256) row[i] = expf(row[i] - rmax) * inv;
}

// Kernel: cond_ctx[b,k,:] = sum_l attn[b,k,l] * V[b,l,:]. k-major B -> VALU kernel.
// ctx is padded to 16 rows per batch (rows 10..15 zero) for the WMMA end stage.
__global__ void condctx_kernel(const float* __restrict__ attn, const float* __restrict__ V,
                               float* __restrict__ ctx) {
  int b = blockIdx.x;
  int n = blockIdx.y * 128 + threadIdx.x;    // 12 * 128 = 1536
  __shared__ float sa[NTOPK][128];
  float acc[NTOPK];
#pragma unroll
  for (int k = 0; k < NTOPK; k++) acc[k] = 0.f;
  for (int l0 = 0; l0 < LL; l0 += 128) {
    __syncthreads();
    for (int t = threadIdx.x; t < NTOPK * 128; t += 128) {
      int k = t / 128, l = t % 128;
      sa[k][l] = attn[((size_t)(b * NTOPK + k)) * LL + l0 + l];
    }
    __syncthreads();
    for (int l = 0; l < 128; l++) {
      float v = V[((size_t)(b * LL + l0 + l)) * EE + n];
#pragma unroll
      for (int k = 0; k < NTOPK; k++) acc[k] += sa[k][l] * v;
    }
  }
#pragma unroll
  for (int k = 0; k < NTOPK; k++) ctx[((size_t)(b * 16 + k)) * EE + n] = acc[k];
  for (int k = NTOPK; k < 16; k++) ctx[((size_t)(b * 16 + k)) * EE + n] = 0.f;
}

// Kernel: end_logits = sum_k w_k * mask(scale * ctx @ token_cmp^T). WMMA + reduce.
__global__ void end_kernel(const float* __restrict__ ctx, const float* __restrict__ Cm,
                           const float* __restrict__ twt, const int* __restrict__ amask,
                           float* __restrict__ endlog) {
  int b = blockIdx.x;
  int lane = threadIdx.x & 31, wave = threadIdx.x >> 5;
  int nt0 = blockIdx.y * 16 + wave * 2;
  int mrow = lane & 15, hi = lane >> 4;
  const float* arow = ctx + (size_t)(b * 16 + mrow) * EE;  // padded rows 10..15 = 0
  v8f acc[2] = {};
  for (int kt = 0; kt < 48; kt++) {
    v16bf ah, al; load_frag_a(arow, kt * 32, hi, ah, al);
#pragma unroll
    for (int j = 0; j < 2; j++) {
      int tok = (nt0 + j) * 16 + mrow;
      const float* crow = Cm + ((size_t)b * LL + tok) * EE;
      v16bf bh, bl; load_frag_b(crow, kt * 32, hi, bh, bl);
      acc[j] = mma3(ah, al, bh, bl, acc[j]);
    }
  }
#pragma unroll
  for (int j = 0; j < 2; j++) {
    float s = 0.f;
#pragma unroll
    for (int p = 0; p < 8; p++) {
      int m = hi * 8 + p;
      float w = (m < NTOPK) ? twt[b * NTOPK + m] : 0.f;
      s += acc[j][p] * w;
    }
    s *= SCALE_E;
    s += __shfl_xor(s, 16, 32);                 // combine the two half-lane row groups
    int tok = (nt0 + j) * 16 + (lane & 15);
    if (hi == 0) {
      bool valid = amask[(size_t)b * LL + tok] != 0;
      endlog[(size_t)b * LL + tok] = valid ? s : MASK_VAL;
    }
  }
}

// ---------------------------------------------------------------------------
extern "C" void kernel_launch(void* const* d_in, const int* in_sizes, int n_in,
                              void* d_out, int out_size, void* d_ws, size_t ws_size,
                              hipStream_t stream) {
  (void)in_sizes; (void)n_in; (void)out_size; (void)ws_size;
  const float* H      = (const float*)d_in[0];
  const int*   amask  = (const int*)d_in[1];
  const float* W_start= (const float*)d_in[2];
  const float* b_start= (const float*)d_in[3];
  const float* W_q    = (const float*)d_in[4];
  const float* b_q    = (const float*)d_in[5];
  const float* W_k    = (const float*)d_in[6];
  const float* b_k    = (const float*)d_in[7];
  const float* W_v    = (const float*)d_in[8];
  const float* b_v    = (const float*)d_in[9];
  const float* W_c    = (const float*)d_in[10];
  const float* b_c    = (const float*)d_in[11];
  float* out = (float*)d_out;

  // Workspace carve-up (~325 MB peak; KVC buffer reused for K_, V, token_cmp).
  char* base = (char*)d_ws; size_t off = 0;
  auto take = [&](size_t bytes) -> char* {
    char* p = base + off; off = (off + bytes + 255) & ~(size_t)255; return p;
  };
  unsigned short* Ahi = (unsigned short*)take((size_t)32768 * DD * 2);
  unsigned short* Alo = (unsigned short*)take((size_t)32768 * DD * 2);
  unsigned short* Whi = (unsigned short*)take((size_t)4 * 2304 * 512 * 2);
  unsigned short* Wlo = (unsigned short*)take((size_t)4 * 2304 * 512 * 2);
  float* KVC  = (float*)take((size_t)32768 * EE * 4);
  float* Qbuf = (float*)take((size_t)80 * EE * 4);
  float* attn = (float*)take((size_t)80 * LL * 4);
  float* ctx  = (float*)take((size_t)BB * 16 * EE * 4);
  int*   tidx = (int*)take(80 * 4);
  float* twt  = (float*)take(80 * 4);
  const size_t WSTRIDE = (size_t)2304 * 512;

  // Prep: repack H and weights into WMMA fragment order (hi/lo bf16).
  repack_H_kernel<<<6144, 256, 0, stream>>>(H, Ahi, Alo);                 // 2048*24*32 / 256
  repack_W_kernel<<<1152, 256, 0, stream>>>(W_q, W_k, W_v, W_c, Whi, Wlo);// 4*2304*32 / 256

  // Stage 1: start logits + top-k.
  start_kernel<<<4096, 256, 0, stream>>>(H, amask, W_start, b_start, out);
  topk_kernel<<<BB, 256, 0, stream>>>(out, tidx, twt);

  // Stage 2: Q projection of gathered top-k rows.
  gemmQ_kernel<<<dim3(5, 6), 256, 0, stream>>>(H, tidx, Whi + 0 * WSTRIDE, Wlo + 0 * WSTRIDE,
                                               b_q, Qbuf);

  // K_ projection -> attention scores -> softmax.
  gemm_big_kernel<<<dim3(512, 12), 256, 0, stream>>>(Ahi, Alo, Whi + 1 * WSTRIDE,
                                                     Wlo + 1 * WSTRIDE, b_k, KVC);
  scores_kernel<<<dim3(BB, 16), 256, 0, stream>>>(Qbuf, KVC, amask, attn);
  attn_softmax_kernel<<<80, 256, 0, stream>>>(attn);

  // V projection (reuses KVC) -> conditional context.
  gemm_big_kernel<<<dim3(512, 12), 256, 0, stream>>>(Ahi, Alo, Whi + 2 * WSTRIDE,
                                                     Wlo + 2 * WSTRIDE, b_v, KVC);
  condctx_kernel<<<dim3(BB, 12), 128, 0, stream>>>(attn, KVC, ctx);

  // token_cmp projection (reuses KVC) -> weighted end logits.
  gemm_big_kernel<<<dim3(512, 12), 256, 0, stream>>>(Ahi, Alo, Whi + 3 * WSTRIDE,
                                                     Wlo + 3 * WSTRIDE, b_c, KVC);
  end_kernel<<<dim3(BB, 16), 256, 0, stream>>>(ctx, KVC, twt, amask, out + (size_t)BB * LL);
}
